// MemoryLayer_32272384262801
// MI455X (gfx1250) — matrix operand order repaired and checked by reading, not compiled, not run on hardware
//
#include <hip/hip_runtime.h>
#include <hip/hip_bf16.h>

// ---------------------------------------------------------------------------
// MemoryLayer on MI455X (gfx1250): bf16 WMMA everywhere, fp32 accumulate.
// B=2 L=512 D=1024 H=16 HD=64 M=16384 TOPK=8.  N = B*L = 1024 token rows.
// Compute-bound (~62 GFLOP vs ~50 MB HBM traffic) -> bf16 matrix pipes.
// CDNA5 paths: v_wmma_f32_16x16x32_bf16, global_load_async_to_lds_b128
// (ASYNCcnt), s_wait_dscnt/s_wait_asynccnt split counters, global_prefetch_b8.
// ---------------------------------------------------------------------------

typedef __attribute__((ext_vector_type(16))) __bf16 v16bf;
typedef __attribute__((ext_vector_type(8)))  float  v8f;

#define WMMA_BF16(A, Bm, C) \
    __builtin_amdgcn_wmma_f32_16x16x32_bf16(false, (A), false, (Bm), (short)0, (C), false, false)

#define CB  2
#define CL  512
#define CD  1024
#define CH  16
#define CHD 64
#define CM  16384
#define CTOPK 8
#define CN  (CB*CL)          // 1024 token rows
#define CLM (CL*CTOPK)       // 4096 memory-prefix length

// ---------------------------------------------------------------------------
// Fragment loaders; A and B use the same contiguous k order (k = half*16 + e).
// A (16xK=32, MxK): lane 0-15 -> rows 0-15 ; half=lane>>4 selects K sub-chunk.
//   elems 0..15 = K[half*16 .. half*16+15]
// B (K=32 x 16, KxN): lane&15 = column ; elems 0..15 = K[half*16 .. half*16+15]
// C/D (16x16 f32): vgpr r, lane l -> row r+8*(l>>4), col l&15
// ---------------------------------------------------------------------------
__device__ inline v16bf load_frag_a(const __bf16* base, int row0, int ldm, int k0, int lane)
{
    const int r  = lane & 15;
    const int hh = lane >> 4;
    const __bf16* p = base + (size_t)(row0 + r) * ldm + k0;
    v16bf a;
#pragma unroll
    for (int e = 0; e < 8; ++e) a[e]     = p[hh * 16 + e];
#pragma unroll
    for (int e = 0; e < 8; ++e) a[8 + e] = p[hh * 16 + 8 + e];
    return a;
}

// B fragment when B[k][j] = src[row j][k]  (K contiguous in memory: the "NT" case)
__device__ inline v16bf load_frag_b_rowptr(const __bf16* rowp, int k0, int hh)
{
    const __bf16* p = rowp + k0 + hh * 16;
    v16bf b;
#pragma unroll
    for (int e = 0; e < 16; ++e) b[e] = p[e];
    return b;
}

__device__ inline v16bf load_frag_b_nt(const __bf16* base, int col0, int ldm, int k0, int lane)
{
    const int j  = lane & 15;
    const int hh = lane >> 4;
    return load_frag_b_rowptr(base + (size_t)(col0 + j) * ldm, k0, hh);
}

// ---------------------------------------------------------------------------
// Kernel 0: fp32 -> bf16 conversion (streaming, bandwidth-trivial: ~50 MB)
// ---------------------------------------------------------------------------
__global__ __launch_bounds__(256) void ml_cvt_bf16(const float* __restrict__ s,
                                                   __bf16* __restrict__ d, int n)
{
    int i  = blockIdx.x * 256 + threadIdx.x;
    int st = gridDim.x * 256;
    for (; i < n; i += st) d[i] = (__bf16)s[i];
}

// ---------------------------------------------------------------------------
// Kernel 1: fused Q/K/V projection.  out[n,j] = sum_d x[n,d] * W[j,d] + b[j]
// 12288 tiles of 16x16 (64 row-tiles x 192 col-tiles across q|k|v).
// ---------------------------------------------------------------------------
__global__ __launch_bounds__(256)
void ml_qkv_gemm(const __bf16* __restrict__ xb,
                 const __bf16* __restrict__ wqb, const __bf16* __restrict__ wkb,
                 const __bf16* __restrict__ wvb,
                 const float* __restrict__ bq, const float* __restrict__ bk,
                 const float* __restrict__ bv,
                 __bf16* __restrict__ q, __bf16* __restrict__ k, __bf16* __restrict__ v)
{
    const int lane = threadIdx.x & 31;
    const int wave = threadIdx.x >> 5;
    const int tile = blockIdx.x * 8 + wave;        // 1536 blocks * 8 waves = 12288
    const int rowT = tile & 63;
    const int colT = tile >> 6;                    // 0..191
    const int which = colT >> 6;                   // 0=q 1=k 2=v
    const int col0 = (colT & 63) * 16;
    const int row0 = rowT * 16;
    const int n16 = lane & 15, hh = lane >> 4;

    const __bf16* w    = (which == 0) ? wqb : (which == 1) ? wkb : wvb;
    const float*  bias = (which == 0) ? bq  : (which == 1) ? bk  : bv;
    __bf16*       outp = (which == 0) ? q   : (which == 1) ? k   : v;

    const float bj = bias[col0 + n16];
    v8f c = {};
#pragma unroll
    for (int r = 0; r < 8; ++r) c[r] = bj;

    for (int k0 = 0; k0 < CD; k0 += 32) {
        v16bf a  = load_frag_a(xb, row0, CD, k0, lane);
        v16bf bf = load_frag_b_nt(w, col0, CD, k0, lane);
        __builtin_prefetch(w + (size_t)(col0 + n16) * CD + k0 + 128, 0, 1);
        c = WMMA_BF16(a, bf, c);
    }
#pragma unroll
    for (int r = 0; r < 8; ++r)
        outp[(size_t)(row0 + r + 8 * hh) * CD + col0 + n16] = (__bf16)c[r];
}

// ---------------------------------------------------------------------------
// Kernel 2: fused similarity GEMM + top-8.   q_flat[16384x64] vs keys[16384x64].
// One block = 16 queries. Each wave WMMAs 16x16 score tiles for its strided set
// of key tiles; each lane keeps a private top-8 for (row = lane&15,
// col-slice = (lane>>4)*8..+7); final LDS merge of 128 candidates per query.
// The 1 GiB sim matrix is never materialized.
// ---------------------------------------------------------------------------
__global__ __launch_bounds__(256)
void ml_sim_topk(const __bf16* __restrict__ qb, const __bf16* __restrict__ mkb,
                 int* __restrict__ tki)
{
    __shared__ float stage[8][16][17];
    __shared__ float candv[16][128];
    __shared__ int   candi[16][128];

    const int lane = threadIdx.x & 31;
    const int wave = threadIdx.x >> 5;
    const int n16 = lane & 15, hh = lane >> 4;
    const int q0 = blockIdx.x * 16;                 // 1024 blocks

    v16bf a0 = load_frag_a(qb, q0, CHD, 0, lane);
    v16bf a1 = load_frag_a(qb, q0, CHD, 32, lane);

    float bestv[8];
    int   besti[8];
#pragma unroll
    for (int e = 0; e < 8; ++e) { bestv[e] = -3.0e38f; besti[e] = 0; }

    for (int t = wave; t < CM / 16; t += 8) {       // 128 key tiles per wave
        const int key0 = t * 16;
        v16bf b0 = load_frag_b_nt(mkb, key0, CHD, 0, lane);
        v16bf b1 = load_frag_b_nt(mkb, key0, CHD, 32, lane);
        v8f c = {};
        c = WMMA_BF16(a0, b0, c);
        c = WMMA_BF16(a1, b1, c);
#pragma unroll
        for (int r = 0; r < 8; ++r) stage[wave][r + 8 * hh][n16] = c[r];
        asm volatile("s_wait_dscnt 0" ::: "memory"); // in-wave LDS RAW (cross-lane)
#pragma unroll
        for (int e = 0; e < 8; ++e) {
            float sv = stage[wave][n16][hh * 8 + e];
            int   gi = key0 + hh * 8 + e;
            if (sv > bestv[7]) {                     // insertion into sorted top-8
                bestv[7] = sv; besti[7] = gi;
#pragma unroll
                for (int s = 7; s > 0; --s)
                    if (bestv[s] > bestv[s - 1]) {
                        float tv = bestv[s]; bestv[s] = bestv[s - 1]; bestv[s - 1] = tv;
                        int   ti = besti[s]; besti[s] = besti[s - 1]; besti[s - 1] = ti;
                    }
            }
        }
    }
    const int slot = (wave * 2 + hh) * 8;           // 16 contributors x 8 per query
#pragma unroll
    for (int e = 0; e < 8; ++e) { candv[n16][slot + e] = bestv[e]; candi[n16][slot + e] = besti[e]; }
    __syncthreads();

    if (threadIdx.x < 16) {
        const int qr = threadIdx.x;
        for (int kk = 0; kk < CTOPK; ++kk) {
            float best = -3.0e38f; int bj = 0;
            for (int j = 0; j < 128; ++j) {
                float vv = candv[qr][j];
                if (vv > best) { best = vv; bj = j; }
            }
            tki[(size_t)(q0 + qr) * CTOPK + kk] = candi[qr][bj];
            candv[qr][bj] = -3.0e38f;
        }
    }
}

// ---------------------------------------------------------------------------
// Kernel 3: flash attention over [4096 memory prefix | 512 causal local].
// One wave = one (b,h,16-query) tile; 32 keys per step; online softmax with
// per-lane stats for the 8 C-rows it owns.  P is re-laid out C->A through LDS.
// V rows (gathered via top-k indices) stream memory->LDS with
// global_load_async_to_lds_b128, overlapping the QK^T WMMAs; we block on
// ASYNCcnt only right before the P*V WMMAs.
// ---------------------------------------------------------------------------
__device__ inline const __bf16* ml_row_ptr(int m, int b, int h,
                                           const __bf16* memtab, const __bf16* loctab,
                                           const int* __restrict__ tki)
{
    if (m < CLM) {
        const int lp = m >> 3, t = m & 7;
        const int fq = (b * CL + lp) * CH + h;
        const int id = tki[(size_t)fq * CTOPK + t];
        return memtab + (size_t)id * CHD;
    }
    const int j = m - CLM;
    return loctab + (size_t)(b * CL + j) * CD + h * CHD;
}

// one async 16B memory->LDS transfer per lane; inst offset applies to BOTH the
// LDS and the global address (ISA 10.x async pseudocode)
#define ML_ASYNC_B128(LDS32, GA64, OFF) \
    asm volatile("global_load_async_to_lds_b128 %0, %1, off offset:" #OFF \
                 :: "v"(LDS32), "v"(GA64) : "memory")

__global__ __launch_bounds__(256)
void ml_attention(const __bf16* __restrict__ qb, const __bf16* __restrict__ kb,
                  const __bf16* __restrict__ vb, const __bf16* __restrict__ mkb,
                  const __bf16* __restrict__ mvb, const int* __restrict__ tki,
                  __bf16* __restrict__ aob)
{
    __shared__ float  pstage[8][16][32];
    __shared__ __bf16 vstage[8][32][64];

    const int lane = threadIdx.x & 31;
    const int wave = threadIdx.x >> 5;
    const int n16 = lane & 15, hh = lane >> 4;

    const int qt = blockIdx.x * 8 + wave;           // 0..1023
    const int lt = qt & 31;
    const int h  = (qt >> 5) & 15;
    const int b  = qt >> 9;

    const int qrow0 = b * CL + lt * 16;
    v16bf qa0 = load_frag_a(qb + h * CHD, qrow0, CD, 0, lane);
    v16bf qa1 = load_frag_a(qb + h * CHD, qrow0, CD, 32, lane);

    // LDS byte address of this lane's vstage row (LDS aperture: offset in addr[31:0])
    const unsigned lds_v = (unsigned)(unsigned long long)(uintptr_t)&vstage[wave][lane][0];

    float mrow[8], lrow[8];
    v8f   oc[4];
    v8f   vzero = {};
#pragma unroll
    for (int r = 0; r < 8; ++r) { mrow[r] = -1.0e30f; lrow[r] = 0.0f; }
#pragma unroll
    for (int ct = 0; ct < 4; ++ct) oc[ct] = vzero;

    // memory tiles (128) + only the local tiles at/below the causal frontier
    const int nkt = 128 + (((lt + 1) * 16 + 31) >> 5);

    for (int kt = 0; kt < nkt; ++kt) {
        const int m0 = kt * 32;

        // previous iteration's LDS column reads must retire before the async
        // engine overwrites vstage (DScnt and ASYNCcnt are unordered)
        asm volatile("s_wait_dscnt 0" ::: "memory");

        // ---- kick off async gather of this step's 32 V rows into LDS
        {
            const __bf16* vr = ml_row_ptr(m0 + lane, b, h, mvb, vb, tki);
            const unsigned long long ga = (unsigned long long)(uintptr_t)vr;
            ML_ASYNC_B128(lds_v, ga, 0);
            ML_ASYNC_B128(lds_v, ga, 16);
            ML_ASYNC_B128(lds_v, ga, 32);
            ML_ASYNC_B128(lds_v, ga, 48);
            ML_ASYNC_B128(lds_v, ga, 64);
            ML_ASYNC_B128(lds_v, ga, 80);
            ML_ASYNC_B128(lds_v, ga, 96);
            ML_ASYNC_B128(lds_v, ga, 112);
        }

        // ---- scores: 16 x 32 in two 16-key subtiles, K=64 as two bf16 WMMAs
        v8f sc[2];
#pragma unroll
        for (int s = 0; s < 2; ++s) {
            const int mcol = m0 + s * 16 + n16;
            const __bf16* kr = ml_row_ptr(mcol, b, h, mkb, kb, tki);
            v16bf b0 = load_frag_b_rowptr(kr, 0, hh);
            v16bf b1 = load_frag_b_rowptr(kr, 32, hh);
            v8f c = vzero;
            c = WMMA_BF16(qa0, b0, c);
            c = WMMA_BF16(qa1, b1, c);
            sc[s] = c;
        }
        // ---- scale + causal mask + online softmax (per-lane stats for 8 rows)
        float oscale[8];
#pragma unroll
        for (int r = 0; r < 8; ++r) {
            const int qpos = lt * 16 + r + 8 * hh;
#pragma unroll
            for (int s = 0; s < 2; ++s) {
                const int m = m0 + s * 16 + n16;
                float v = sc[s][r] * 0.125f;        // 1/sqrt(64)
                if (m >= CLM && (m - CLM) > qpos) v = -1.0e30f;
                sc[s][r] = v;
            }
            float tmax = fmaxf(sc[0][r], sc[1][r]);
#pragma unroll
            for (int off = 1; off < 16; off <<= 1)
                tmax = fmaxf(tmax, __shfl_xor(tmax, off, 32));
            const float mn = fmaxf(mrow[r], tmax);
            const float sc8 = __expf(mrow[r] - mn);
            const float p0 = __expf(sc[0][r] - mn);
            const float p1 = __expf(sc[1][r] - mn);
            float ts = p0 + p1;
#pragma unroll
            for (int off = 1; off < 16; off <<= 1) ts += __shfl_xor(ts, off, 32);
            lrow[r] = lrow[r] * sc8 + ts;
            mrow[r] = mn;
            oscale[r] = sc8;
            const int M = r + 8 * hh;
            pstage[wave][M][n16]      = p0;
            pstage[wave][M][16 + n16] = p1;
        }
#pragma unroll
        for (int ct = 0; ct < 4; ++ct)
#pragma unroll
            for (int r = 0; r < 8; ++r) oc[ct][r] *= oscale[r];

        asm volatile("s_wait_dscnt 0" ::: "memory");    // pstage stores visible

        // ---- re-read P in the bf16 A layout (16 rows x K=32 keys)
        v16bf pa;
#pragma unroll
        for (int e = 0; e < 8; ++e) pa[e]     = (__bf16)pstage[wave][n16][hh * 16 + e];
#pragma unroll
        for (int e = 0; e < 8; ++e) pa[8 + e] = (__bf16)pstage[wave][n16][hh * 16 + 8 + e];

        asm volatile("s_wait_asynccnt 0" ::: "memory"); // vstage async writes landed

        // ---- O += P x V : B[k][j] = vstage[k][ct*16+j]  (column reads)
#pragma unroll
        for (int ct = 0; ct < 4; ++ct) {
            v16bf vbf;
#pragma unroll
            for (int e = 0; e < 16; ++e) vbf[e] = vstage[wave][hh * 16 + e][ct * 16 + n16];
            oc[ct] = WMMA_BF16(pa, vbf, oc[ct]);
        }
    }
    // ---- normalize and store attention output (bf16, row-major [1024 x 1024])
#pragma unroll
    for (int ct = 0; ct < 4; ++ct)
#pragma unroll
        for (int r = 0; r < 8; ++r) {
            const int M = r + 8 * hh;
            aob[(size_t)(b * CL + lt * 16 + M) * CD + h * CHD + ct * 16 + n16] =
                (__bf16)(oc[ct][r] / lrow[r]);
        }
}

// ---------------------------------------------------------------------------
// Kernel 4: output projection, fp32 result.
// ---------------------------------------------------------------------------
__global__ __launch_bounds__(256)
void ml_out_gemm(const __bf16* __restrict__ ab, const __bf16* __restrict__ wob,
                 const float* __restrict__ bo, float* __restrict__ out)
{
    const int lane = threadIdx.x & 31;
    const int wave = threadIdx.x >> 5;
    const int tile = blockIdx.x * 8 + wave;         // 512 blocks * 8 = 4096 tiles
    const int rowT = tile & 63;
    const int colT = tile >> 6;
    const int col0 = colT * 16, row0 = rowT * 16;
    const int n16 = lane & 15, hh = lane >> 4;

    const float bj = bo[col0 + n16];
    v8f c = {};
#pragma unroll
    for (int r = 0; r < 8; ++r) c[r] = bj;

    for (int k0 = 0; k0 < CD; k0 += 32) {
        v16bf a  = load_frag_a(ab, row0, CD, k0, lane);
        v16bf bf = load_frag_b_nt(wob, col0, CD, k0, lane);
        __builtin_prefetch(wob + (size_t)(col0 + n16) * CD + k0 + 128, 0, 1);
        c = WMMA_BF16(a, bf, c);
    }
#pragma unroll
    for (int r = 0; r < 8; ++r)
        out[(size_t)(row0 + r + 8 * hh) * CD + col0 + n16] = c[r];
}

// ---------------------------------------------------------------------------
extern "C" void kernel_launch(void* const* d_in, const int* in_sizes, int n_in,
                              void* d_out, int out_size, void* d_ws, size_t ws_size,
                              hipStream_t stream)
{
    const float* x  = (const float*)d_in[0];
    const float* mk = (const float*)d_in[1];
    const float* mv = (const float*)d_in[2];
    const float* Wq = (const float*)d_in[3];
    const float* bq = (const float*)d_in[4];
    const float* Wk = (const float*)d_in[5];
    const float* bk = (const float*)d_in[6];
    const float* Wv = (const float*)d_in[7];
    const float* bv = (const float*)d_in[8];
    const float* Wo = (const float*)d_in[9];
    const float* bo = (const float*)d_in[10];
    float* out = (float*)d_out;

    char* ws = (char*)d_ws;
    const size_t MB = 1024ull * 1024ull;
    __bf16* xb  = (__bf16*)(ws + 0 * MB);
    __bf16* wqb = (__bf16*)(ws + 2 * MB);
    __bf16* wkb = (__bf16*)(ws + 4 * MB);
    __bf16* wvb = (__bf16*)(ws + 6 * MB);
    __bf16* wob = (__bf16*)(ws + 8 * MB);
    __bf16* mkb = (__bf16*)(ws + 10 * MB);
    __bf16* mvb = (__bf16*)(ws + 12 * MB);
    __bf16* qb  = (__bf16*)(ws + 14 * MB);
    __bf16* kb  = (__bf16*)(ws + 16 * MB);
    __bf16* vb  = (__bf16*)(ws + 18 * MB);
    __bf16* aob = (__bf16*)(ws + 20 * MB);
    int*    tki = (int*)   (ws + 22 * MB);

    const int NE = CN * CD;     // 1M elements per matrix
    ml_cvt_bf16<<<1024, 256, 0, stream>>>(x,  xb,  NE);
    ml_cvt_bf16<<<1024, 256, 0, stream>>>(Wq, wqb, NE);
    ml_cvt_bf16<<<1024, 256, 0, stream>>>(Wk, wkb, NE);
    ml_cvt_bf16<<<1024, 256, 0, stream>>>(Wv, wvb, NE);
    ml_cvt_bf16<<<1024, 256, 0, stream>>>(Wo, wob, NE);
    ml_cvt_bf16<<<1024, 256, 0, stream>>>(mk, mkb, CM * CHD);
    ml_cvt_bf16<<<1024, 256, 0, stream>>>(mv, mvb, CM * CHD);

    ml_qkv_gemm<<<1536, 256, 0, stream>>>(xb, wqb, wkb, wvb, bq, bk, bv, qb, kb, vb);
    ml_sim_topk<<<1024, 256, 0, stream>>>(qb, mkb, tki);
    ml_attention<<<128, 256, 0, stream>>>(qb, kb, vb, mkb, mvb, tki, aob);
    ml_out_gemm<<<512, 256, 0, stream>>>(aob, wob, bo, out);
}
